// WindowAttention_31705448579784
// MI455X (gfx1250) — matrix operand order repaired and checked by reading, not compiled, not run on hardware
//
#include <hip/hip_runtime.h>
#include <math.h>

// ---------------- constants ----------------
#define DIMC   256
#define NTOK   64
#define NHEAD  8
#define HD     16
#define QSCALE 0.25f   // HD^-0.5

typedef __bf16 bf16;
typedef __attribute__((ext_vector_type(16))) __bf16 v16bf;
typedef __attribute__((ext_vector_type(8)))  float  v8f;

// ---------------- LDS layout (bytes), total 256KB ----------------
// Phase 1/2:  [0,64K) x fp32 ; [64K,160K) yq/yk/yv bf16 [px][256]
// Phase 2 out:[160K,192K) Q bf16 [tok][256]; [192K,208K) K bf16 [tok][128];
//             [208K,224K) V^T bf16 [128][64]
// Phase 3:    reuse [0,128K) as 8x16KB fp32 score scratch;
//             [128K,132K) stats; [132K,148K) staged mask fp32
// Phase 3 out:[224K,256K) O bf16 [tok][256]
#define OFF_X    0
#define OFF_YQ   65536
#define OFF_YK   98304
#define OFF_YV   131072
#define OFF_Q    163840
#define OFF_K    196608
#define OFF_VT   212992
#define OFF_SCR  0
#define OFF_STAT 131072
#define OFF_MASK 135168
#define OFF_O    229376
#define SMEM_BYTES 262144

// ws layout: bf16 weights + expanded rpb (L2-resident, reused by all blocks)
#define WS_WQ  0        // 256x256 bf16
#define WS_WK  131072   // 128x256 bf16 (only first half of K used: k2=k1)
#define WS_WV  196608   // 128x256 bf16 (only first half of V used: v2=v1)
#define WS_WP  262144   // 256x256 bf16 (proj_w row-major [co'][ci])
#define WS_RPB 393216   // 8x64x64 f32, rpb_full[h][n][m]

__device__ __forceinline__ v8f wmma_bf16(v16bf a, v16bf b, v8f c) {
  // D = A(16x32 bf16) * B(32x16 bf16) + C(16x16 f32)
  return __builtin_amdgcn_wmma_f32_16x16x32_bf16(false, a, false, b,
                                                 (short)0, c, false, false);
}

// CDNA5 async global->LDS copy (ASYNCcnt-tracked, no VGPR round trip).
// LDS destination offset comes from the vdst VGPR; per-lane 16B transfer.
__device__ __forceinline__ void async_g2l_b128(unsigned lds_off,
                                               unsigned long long gaddr) {
  asm volatile("global_load_async_to_lds_b128 %0, %1, off"
               :: "v"(lds_off), "v"(gaddr) : "memory");
}
__device__ __forceinline__ void wait_async0() {
  asm volatile("s_wait_asynccnt 0" ::: "memory");
}

// A fragment from M-major bf16 buffer (row stride = `stride` elements).
// Lane layout (ISA 7.12.2): M = lane&15; lanes<16 hold K = k0+{0..7,16..23},
// lanes>=16 hold K = k0+{8..15,24..31}. Chunks beyond klimit are zero-padded.
__device__ __forceinline__ v16bf load_A(const bf16* base, int stride,
                                        int M0, int k0, int klimit, int lane) {
  const int m  = M0 + (lane & 15);
  const int hi = (lane >> 4) << 3;           // 0 or 8
  const bf16* row = base + (size_t)m * stride;
  const int kA = k0 + hi;
  const int kB = k0 + 16 + hi;
  v16bf a;
#pragma unroll
  for (int i = 0; i < 8; ++i) a[i]     = (kA < klimit) ? row[kA + i] : (bf16)0.0f;
#pragma unroll
  for (int i = 0; i < 8; ++i) a[i + 8] = (kB < klimit) ? row[kB + i] : (bf16)0.0f;
  return a;
}

// B fragment from N-major bf16 buffer (row = one N column, contiguous in K).
// Lane layout: N = lane&15; lanes<16 hold K = k0+{0..7,16..23}, lanes>=16 K = k0+{8..15,24..31}.
__device__ __forceinline__ v16bf load_B(const bf16* base, int stride,
                                        int N0, int k0, int klimit, int lane) {
  const int n  = N0 + (lane & 15);
  const int kA = k0 + ((lane >> 4) << 3);    // 0 or 8
  const int kB = kA + 16;
  const bf16* p = base + (size_t)n * stride;
  v16bf b;
#pragma unroll
  for (int i = 0; i < 8; ++i) b[i]     = (kA < klimit) ? p[kA + i] : (bf16)0.0f;
#pragma unroll
  for (int i = 0; i < 8; ++i) b[i + 8] = (kB < klimit) ? p[kB + i] : (bf16)0.0f;
  return b;
}

extern "C" __global__ void prep_weights(const float* __restrict__ pw_q,
                                        const float* __restrict__ pw_k,
                                        const float* __restrict__ pw_v,
                                        const float* __restrict__ proj_w,
                                        const float* __restrict__ rpb_table,
                                        bf16* wq, bf16* wk, bf16* wv, bf16* wp,
                                        float* rpbf) {
  int gid = blockIdx.x * blockDim.x + threadIdx.x;
  if      (gid <  65536) wq[gid]          = (bf16)pw_q[gid];
  else if (gid <  98304) wk[gid -  65536] = (bf16)pw_k[gid - 65536];
  else if (gid < 131072) wv[gid -  98304] = (bf16)pw_v[gid - 98304];
  else if (gid < 196608) wp[gid - 131072] = (bf16)proj_w[gid - 131072];
  else if (gid < 229376) {
    // expand relative-position bias: rpb_full[h][n][m]
    const int e = gid - 196608;
    const int h = e >> 12;
    const int n = (e >> 6) & 63, m = e & 63;
    const int ri = ((n >> 3) - (m >> 3) + 7) * 15 + ((n & 7) - (m & 7) + 7);
    rpbf[e] = rpb_table[ri * NHEAD + h];
  }
}

extern "C" __global__ __launch_bounds__(256)
void win_attn_fused(const float* __restrict__ x, const float* __restrict__ mask,
                    const float* __restrict__ rpb_full,
                    const float* __restrict__ proj_b,
                    const float* __restrict__ dw_q, const float* __restrict__ db_q,
                    const float* __restrict__ pb_q,
                    const float* __restrict__ dw_k, const float* __restrict__ db_k,
                    const float* __restrict__ pb_k,
                    const float* __restrict__ dw_v, const float* __restrict__ db_v,
                    const float* __restrict__ pb_v,
                    const bf16* __restrict__ wq, const bf16* __restrict__ wk,
                    const bf16* __restrict__ wv, const bf16* __restrict__ wp,
                    float* __restrict__ out) {
  extern __shared__ __align__(16) char smem[];
  const int b    = blockIdx.x;
  const int tid  = threadIdx.x;
  const int lane = tid & 31;     // wave32
  const int wave = tid >> 5;     // 8 waves
  // Low 32 bits of the LDS-aperture flat address == wave-relative LDS offset.
  const unsigned smem_base = (unsigned)(uintptr_t)smem;

  float* xs = (float*)(smem + OFF_X);
  bf16*  yq = (bf16*)(smem + OFF_YQ);
  bf16*  yk = (bf16*)(smem + OFF_YK);
  bf16*  yv = (bf16*)(smem + OFF_YV);
  bf16*  Qb = (bf16*)(smem + OFF_Q);
  bf16*  Kb = (bf16*)(smem + OFF_K);
  bf16*  Vt = (bf16*)(smem + OFF_VT);
  bf16*  Ob = (bf16*)(smem + OFF_O);

  // ---------- Phase 0: async-stage x (64 tok x 256 ch fp32, 64KB) ----------
  {
    const unsigned long long g =
        (unsigned long long)(uintptr_t)(x + (size_t)b * NTOK * DIMC) +
        (unsigned long long)tid * 16;
    const unsigned l = smem_base + OFF_X + tid * 16;
#pragma unroll
    for (int i = 0; i < 16; ++i)
      async_g2l_b128(l + i * 4096, g + (unsigned long long)i * 4096);
    wait_async0();
  }
  __syncthreads();

  // ---------- Phase 1: depthwise 3x3 + bias + ReLU (q,k,v fused) ----------
  {
    const int c = tid;  // one channel per thread
    float w9q[9], w9k[9], w9v[9];
#pragma unroll
    for (int j = 0; j < 9; ++j) {
      w9q[j] = dw_q[c * 9 + j];
      w9k[j] = dw_k[c * 9 + j];
      w9v[j] = dw_v[c * 9 + j];
    }
    const float bq = db_q[c], bk = db_k[c], bv = db_v[c];
    for (int px = 0; px < 64; ++px) {
      const int r = px >> 3, cc = px & 7;
      float aq = 0.f, ak = 0.f, av = 0.f;
#pragma unroll
      for (int ky = 0; ky < 3; ++ky)
#pragma unroll
        for (int kx = 0; kx < 3; ++kx) {
          const int rr = r + ky - 1, c2 = cc + kx - 1;
          if (rr >= 0 && rr < 8 && c2 >= 0 && c2 < 8) {
            const float xv = xs[(rr * 8 + c2) * DIMC + c];
            const int j = ky * 3 + kx;
            aq = fmaf(w9q[j], xv, aq);
            ak = fmaf(w9k[j], xv, ak);
            av = fmaf(w9v[j], xv, av);
          }
        }
      yq[px * DIMC + c] = (bf16)fmaxf(aq + bq, 0.f);
      yk[px * DIMC + c] = (bf16)fmaxf(ak + bk, 0.f);
      yv[px * DIMC + c] = (bf16)fmaxf(av + bv, 0.f);
    }
  }
  __syncthreads();

  // ---------- Phase 2: pointwise 1x1 convs as WMMA GEMMs ----------
  const int hi8 = (lane >> 4) << 3;
  // Q: 16 co-tiles x 4 px-tiles; 8 tiles per wave. Scale folded into store.
  for (int i = 0; i < 8; ++i) {
    const int idx = wave * 8 + i;
    const int cot = idx >> 2, pxt = idx & 3;
    __builtin_prefetch(wq + (size_t)(cot * 16) * 256, 0, 0);
    v8f acc;
#pragma unroll
    for (int r = 0; r < 8; ++r) acc[r] = pb_q[cot * 16 + r + hi8];
    for (int k0 = 0; k0 < 256; k0 += 32)
      acc = wmma_bf16(load_A(wq, 256, cot * 16, k0, 256, lane),
                      load_B(yq, 256, pxt * 16, k0, 256, lane), acc);
    const int px = pxt * 16 + (lane & 15);
#pragma unroll
    for (int r = 0; r < 8; ++r)
      Qb[px * DIMC + cot * 16 + r + hi8] = (bf16)(acc[r] * QSCALE);
  }
  // K (first 128 out channels only): 8x4 tiles; 4 per wave.
  for (int i = 0; i < 4; ++i) {
    const int idx = wave * 4 + i;
    const int cot = idx >> 2, pxt = idx & 3;
    __builtin_prefetch(wk + (size_t)(cot * 16) * 256, 0, 0);
    v8f acc;
#pragma unroll
    for (int r = 0; r < 8; ++r) acc[r] = pb_k[cot * 16 + r + hi8];
    for (int k0 = 0; k0 < 256; k0 += 32)
      acc = wmma_bf16(load_A(wk, 256, cot * 16, k0, 256, lane),
                      load_B(yk, 256, pxt * 16, k0, 256, lane), acc);
    const int px = pxt * 16 + (lane & 15);
#pragma unroll
    for (int r = 0; r < 8; ++r)
      Kb[px * 128 + cot * 16 + r + hi8] = (bf16)acc[r];
  }
  // V (first 128 out channels only), stored transposed [ch][tok].
  for (int i = 0; i < 4; ++i) {
    const int idx = wave * 4 + i;
    const int cot = idx >> 2, pxt = idx & 3;
    __builtin_prefetch(wv + (size_t)(cot * 16) * 256, 0, 0);
    v8f acc;
#pragma unroll
    for (int r = 0; r < 8; ++r) acc[r] = pb_v[cot * 16 + r + hi8];
    for (int k0 = 0; k0 < 256; k0 += 32)
      acc = wmma_bf16(load_A(wv, 256, cot * 16, k0, 256, lane),
                      load_B(yv, 256, pxt * 16, k0, 256, lane), acc);
    const int px = pxt * 16 + (lane & 15);
#pragma unroll
    for (int r = 0; r < 8; ++r)
      Vt[(cot * 16 + r + hi8) * 64 + px] = (bf16)acc[r];
  }
  __syncthreads();

  // ---------- Phase 3a: async-stage this window's mask (16KB) ----------
  {
    const unsigned long long g =
        (unsigned long long)(uintptr_t)(mask + (size_t)b * NTOK * NTOK) +
        (unsigned long long)tid * 16;
    const unsigned l = smem_base + OFF_MASK + tid * 16;
#pragma unroll
    for (int i = 0; i < 4; ++i)
      async_g2l_b128(l + i * 4096, g + (unsigned long long)i * 4096);
    wait_async0();
  }
  __syncthreads();

  // ---------- Phase 3b: dual-half attention, one head per wave ----------
  {
    const int h = wave;
    float* Swv  = (float*)(smem + OFF_SCR) + wave * (64 * 64);
    float* stat = (float*)(smem + OFF_STAT) + wave * 128;  // [rmax64|rinv64]
    const float* msk  = (const float*)(smem + OFF_MASK);
    const float* rpbh = rpb_full + (size_t)h * (64 * 64);
#pragma unroll 1
    for (int s = 0; s < 2; ++s) {
      const int qoff = (s == 0) ? h * HD : 128 + h * HD;
      // S = Q_h K_h^T + rpb + mask (bias preloaded into accumulator C)
      for (int tr = 0; tr < 4; ++tr)
        for (int tc = 0; tc < 4; ++tc) {
          const int m = tc * 16 + (lane & 15);
          v8f acc;
#pragma unroll
          for (int r = 0; r < 8; ++r) {
            const int n = tr * 16 + r + hi8;
            acc[r] = rpbh[n * 64 + m] + msk[n * 64 + m];
          }
          acc = wmma_bf16(load_A(Qb + qoff, DIMC, tr * 16, 0, 16, lane),
                          load_B(Kb + h * HD, 128, tc * 16, 0, 16, lane), acc);
#pragma unroll
          for (int r = 0; r < 8; ++r)
            Swv[(tr * 16 + r + hi8) * 64 + m] = acc[r];
        }
      // Row stats (2 rows per lane): max and 1/sum(exp) in fp32.
#pragma unroll
      for (int rr = 0; rr < 2; ++rr) {
        const int row = lane * 2 + rr;
        const float4* rp = (const float4*)(Swv + row * 64);
        float mx = -3.0e38f;
#pragma unroll
        for (int i4 = 0; i4 < 16; ++i4) {
          const float4 v = rp[i4];
          mx = fmaxf(mx, fmaxf(fmaxf(v.x, v.y), fmaxf(v.z, v.w)));
        }
        float sum = 0.f;
#pragma unroll
        for (int i4 = 0; i4 < 16; ++i4) {
          const float4 v = rp[i4];
          sum += __expf(v.x - mx) + __expf(v.y - mx) +
                 __expf(v.z - mx) + __expf(v.w - mx);
        }
        stat[row]      = mx;
        stat[64 + row] = 1.f / sum;
      }
      // O = softmax(S) @ V_h  (P materialized on the fly into A fragments)
      for (int tr = 0; tr < 4; ++tr) {
        v8f acc;
#pragma unroll
        for (int r = 0; r < 8; ++r) acc[r] = 0.f;
        for (int kk = 0; kk < 2; ++kk) {
          const int k0 = kk * 32;
          const int mrow = tr * 16 + (lane & 15);
          const float rmax = stat[mrow], rinv = stat[64 + mrow];
          const float* srow = Swv + mrow * 64;
          v16bf a;
#pragma unroll
          for (int i = 0; i < 8; ++i)
            a[i]     = (bf16)(__expf(srow[k0 + hi8 + i]      - rmax) * rinv);
#pragma unroll
          for (int i = 0; i < 8; ++i)
            a[i + 8] = (bf16)(__expf(srow[k0 + 16 + hi8 + i] - rmax) * rinv);
          acc = wmma_bf16(a, load_B(Vt + (h * HD) * 64, 64, 0, k0, 64, lane), acc);
        }
        const int d = lane & 15;
#pragma unroll
        for (int r = 0; r < 8; ++r)
          Ob[(tr * 16 + r + hi8) * DIMC + qoff + d] = (bf16)acc[r];
      }
    }
  }
  __syncthreads();

  // ---------- Phase 4: output projection (64x256) @ (256x256)^T + b ----------
  for (int i = 0; i < 8; ++i) {
    const int idx  = wave * 8 + i;
    const int tokt = idx & 3, cot = idx >> 2;  // 4 tok-tiles x 16 co-tiles
    __builtin_prefetch(wp + (size_t)(cot * 16) * 256, 0, 0);
    const float bias = proj_b[cot * 16 + (lane & 15)];
    v8f acc;
#pragma unroll
    for (int r = 0; r < 8; ++r) acc[r] = bias;
    for (int k0 = 0; k0 < 256; k0 += 32)
      acc = wmma_bf16(load_A(Ob, DIMC, tokt * 16, k0, 256, lane),
                      load_B(wp, 256, cot * 16, k0, 256, lane), acc);
    const int co = cot * 16 + (lane & 15);
#pragma unroll
    for (int r = 0; r < 8; ++r)
      out[((size_t)b * 64 + tokt * 16 + r + hi8) * DIMC + co] = acc[r];
  }
}

extern "C" void kernel_launch(void* const* d_in, const int* in_sizes, int n_in,
                              void* d_out, int out_size, void* d_ws, size_t ws_size,
                              hipStream_t stream) {
  const float* x         = (const float*)d_in[0];
  const float* mask      = (const float*)d_in[1];
  const float* rpb_table = (const float*)d_in[2];
  const float* proj_w    = (const float*)d_in[3];
  const float* proj_b    = (const float*)d_in[4];
  const float* dw_q = (const float*)d_in[5];
  const float* db_q = (const float*)d_in[6];
  const float* pw_q = (const float*)d_in[7];
  const float* pb_q = (const float*)d_in[8];
  const float* dw_k = (const float*)d_in[9];
  const float* db_k = (const float*)d_in[10];
  const float* pw_k = (const float*)d_in[11];
  const float* pb_k = (const float*)d_in[12];
  const float* dw_v = (const float*)d_in[13];
  const float* db_v = (const float*)d_in[14];
  const float* pw_v = (const float*)d_in[15];
  const float* pb_v = (const float*)d_in[16];
  float* out = (float*)d_out;

  char* ws = (char*)d_ws;
  bf16*  wq   = (bf16*)(ws + WS_WQ);
  bf16*  wk   = (bf16*)(ws + WS_WK);
  bf16*  wv   = (bf16*)(ws + WS_WV);
  bf16*  wp   = (bf16*)(ws + WS_WP);
  float* rpbf = (float*)(ws + WS_RPB);

  prep_weights<<<896, 256, 0, stream>>>(pw_q, pw_k, pw_v, proj_w, rpb_table,
                                        wq, wk, wv, wp, rpbf);
  win_attn_fused<<<1024, 256, SMEM_BYTES, stream>>>(
      x, mask, rpbf, proj_b,
      dw_q, db_q, pb_q, dw_k, db_k, pb_k, dw_v, db_v, pb_v,
      wq, wk, wv, wp, out);
}
